// MultiHeadCausalTensionLayer_50912542327306
// MI455X (gfx1250) — matrix operand order repaired and checked by reading, not compiled, not run on hardware
//
#include <hip/hip_runtime.h>
#include <hip/hip_bf16.h>

// ---------------- constants ----------------
#define TT   2048
#define DD   1024
#define HH   16
#define HD   64
#define WN   64
#define NQKV 3072

typedef __attribute__((ext_vector_type(16))) __bf16 v16bf;
typedef __attribute__((ext_vector_type(8)))  float  v8f;
typedef __attribute__((ext_vector_type(4)))  unsigned int v4u;

union ABPack { v16bf v; v4u u[2]; };

__device__ __forceinline__ unsigned short f2bf(float f) {
    unsigned u = __builtin_bit_cast(unsigned, f);
    u += 0x7fffu + ((u >> 16) & 1u);
    return (unsigned short)(u >> 16);
}

__device__ __forceinline__ v8f wmma_bf16(v16bf a, v16bf b, v8f c) {
    return __builtin_amdgcn_wmma_f32_16x16x32_bf16(
        false, a, false, b, (short)0, c, false, false);
}

// A operand (16x32 bf16, MxK). base = &A[row0][k0], rowStride in elements.
// Lane layout: 16 contiguous K at offset hi*16, row = l15.
__device__ __forceinline__ v16bf loadA16x32(const unsigned short* base, int rowStride,
                                            int l15, int hi) {
    ABPack p;
    const unsigned short* p0 = base + (size_t)l15 * rowStride + hi * 16;
    p.u[0] = *(const v4u*)(p0);
    p.u[1] = *(const v4u*)(p0 + 8);
    return p.v;
}

// B operand loaded from B^T stored row-major [N][K]. base = &BT[n0][k0].
// Lane layout: 16 contiguous K at offset hi*16, row (=N) = l15.
__device__ __forceinline__ v16bf loadBT16(const unsigned short* base, int rowStride,
                                          int l15, int hi) {
    ABPack p;
    const unsigned short* p0 = base + (size_t)l15 * rowStride + hi * 16;
    p.u[0] = *(const v4u*)(p0);
    p.u[1] = *(const v4u*)(p0 + 8);
    return p.v;
}

// ---------------- conversion kernels ----------------
__global__ void cvt_bf16_kernel(const float* __restrict__ in,
                                unsigned short* __restrict__ out, int n) {
    int i = blockIdx.x * blockDim.x + threadIdx.x;
    if (i < n) out[i] = f2bf(in[i]);
}

// out[n][k] = bf16(in[k][n]); both 1024x1024
__global__ void transposeW_kernel(const float* __restrict__ in,
                                  unsigned short* __restrict__ out) {
    __shared__ float tile[32][33];
    int bx = blockIdx.x, by = blockIdx.y;
    int tx = threadIdx.x, ty = threadIdx.y;
#pragma unroll
    for (int i = 0; i < 32; i += 8)
        tile[ty + i][tx] = in[(size_t)(by * 32 + ty + i) * DD + bx * 32 + tx];
    __syncthreads();
#pragma unroll
    for (int i = 0; i < 32; i += 8)
        out[(size_t)(bx * 32 + ty + i) * DD + by * 32 + tx] = f2bf(tile[tx][ty + i]);
}

// ---------------- fused QKV GEMM + RoPE ----------------
// xb:  bf16 [T][D] ; wT: bf16 [3072][1024] (concat wq^T|wk^T|wv^T)
// qb,kb: bf16 [H][T][HD] (roped) ; vT: bf16 [H][HD][T]
__global__ void gemm_qkv_kernel(const unsigned short* __restrict__ xb,
                                const unsigned short* __restrict__ wT,
                                unsigned short* __restrict__ qb,
                                unsigned short* __restrict__ kb,
                                unsigned short* __restrict__ vT) {
    int lane = threadIdx.x & 31;
    int l15 = lane & 15, hi = lane >> 4;
    int gw = blockIdx.x * 8 + (threadIdx.x >> 5);   // 6144 waves
    int mt = gw / 48, ng = gw % 48;
    int m0 = mt * 16, n0 = ng * 64;

    v8f acc[4];
#pragma unroll
    for (int j = 0; j < 4; ++j) acc[j] = (v8f){0,0,0,0,0,0,0,0};

    const unsigned short* abase = xb + (size_t)m0 * DD;
    for (int k0 = 0; k0 < DD; k0 += 32) {
        __builtin_prefetch(abase + (size_t)l15 * DD + k0 + 64, 0, 1);
        v16bf a = loadA16x32(abase + k0, DD, l15, hi);
#pragma unroll
        for (int j = 0; j < 4; ++j) {
            v16bf b = loadBT16(wT + (size_t)(n0 + j * 16) * DD + k0, DD, l15, hi);
            acc[j] = wmma_bf16(a, b, acc[j]);
        }
    }

    int mat = n0 >> 10;
    int h = (n0 & 1023) >> 6;
    if (mat < 2) {
        // RoPE: pairs (d, d+32) live in acc[j] / acc[j+2], d = j*16 + l15 (j=0,1)
#pragma unroll
        for (int j = 0; j < 2; ++j) {
#pragma unroll
            for (int r = 0; r < 8; ++r) {
                int t = m0 + r + hi * 8;
                int d = j * 16 + l15;                       // 0..31
                float ang = (float)t * __expf((float)d * (-9.210340371976184f / 32.0f));
                float sn, cs;
                __sincosf(ang, &sn, &cs);
                float x1 = acc[j][r], x2 = acc[j + 2][r];
                acc[j][r]     = x1 * cs - x2 * sn;
                acc[j + 2][r] = x1 * sn + x2 * cs;
            }
        }
        unsigned short* dst = (mat == 0) ? qb : kb;
#pragma unroll
        for (int j = 0; j < 4; ++j)
#pragma unroll
            for (int r = 0; r < 8; ++r) {
                int t = m0 + r + hi * 8;
                dst[((size_t)h * TT + t) * HD + j * 16 + l15] = f2bf(acc[j][r]);
            }
    } else {
        // V stored transposed: vT[h][d][t]
#pragma unroll
        for (int j = 0; j < 4; ++j)
#pragma unroll
            for (int r = 0; r < 8; ++r) {
                int t = m0 + r + hi * 8;
                vT[((size_t)h * HD + j * 16 + l15) * TT + t] = f2bf(acc[j][r]);
            }
    }
}

// ---------------- sliding-window sigmoid attention ----------------
// one wave per (16-query block, head); msg: bf16 [T][D]
__global__ void attn_kernel(const unsigned short* __restrict__ qb,
                            const unsigned short* __restrict__ kb,
                            const unsigned short* __restrict__ vT,
                            unsigned short* __restrict__ msg) {
    __shared__ __align__(16) unsigned short tauLds[8][16][96];

    int lane = threadIdx.x & 31;
    int l15 = lane & 15, hi = lane >> 4;
    int wid = threadIdx.x >> 5;
    int gw = blockIdx.x * 8 + wid;        // 2048 waves
    int h = gw & 15;
    int t0 = (gw >> 4) * 16;

    const unsigned short* qh = qb + ((size_t)h * TT + t0) * HD;
    v16bf aq0 = loadA16x32(qh + 0,  HD, l15, hi);
    v16bf aq1 = loadA16x32(qh + 32, HD, l15, hi);

    float tausum[8];
#pragma unroll
    for (int r = 0; r < 8; ++r) tausum[r] = 0.0f;

    // scores over 3 chunks of 32 keys (2 sub-tiles of 16 keys each)
#pragma unroll
    for (int c = 0; c < 3; ++c) {
#pragma unroll
        for (int s = 0; s < 2; ++s) {
            int key0 = t0 - WN + c * 32 + s * 16;
            int keyrow = key0 + l15;
            int keyrow_c = min(max(keyrow, 0), TT - 1);
            const unsigned short* krow = kb + ((size_t)h * TT + keyrow_c) * HD;
            ABPack b0, b1;
            {
                const unsigned short* p0 = krow + 0 + hi * 16;
                b0.u[0] = *(const v4u*)(p0); b0.u[1] = *(const v4u*)(p0 + 8);
            }
            {
                const unsigned short* p0 = krow + 32 + hi * 16;
                b1.u[0] = *(const v4u*)(p0); b1.u[1] = *(const v4u*)(p0 + 8);
            }
            v8f sc = (v8f){0,0,0,0,0,0,0,0};
            sc = wmma_bf16(aq0, b0.v, sc);
            sc = wmma_bf16(aq1, b1.v, sc);
#pragma unroll
            for (int r = 0; r < 8; ++r) {
                int t = t0 + r + hi * 8;
                int key = key0 + l15;
                bool valid = (key >= 0) && (key >= t - WN) && (key < t);
                float tau = valid ? 1.0f / (1.0f + __expf(-sc[r] * 0.125f)) : 0.0f;
                tausum[r] += tau;
                tauLds[wid][r + hi * 8][c * 32 + s * 16 + l15] = f2bf(tau);
            }
        }
    }

    // msg = tau @ V (A from LDS tau tile, B^T = vT[h][d][t])
    v8f macc[4];
#pragma unroll
    for (int j = 0; j < 4; ++j) macc[j] = (v8f){0,0,0,0,0,0,0,0};

#pragma unroll
    for (int c = 0; c < 3; ++c) {
        v16bf at = loadA16x32(&tauLds[wid][0][0] + c * 32, 96, l15, hi);
        int k0 = t0 - WN + c * 32;
        int col = k0 + hi * 16;
        int colc = min(max(col, 0), TT - 16);
#pragma unroll
        for (int j = 0; j < 4; ++j) {
            const unsigned short* vp =
                vT + ((size_t)h * HD + j * 16 + l15) * TT + colc;
            ABPack bp;
            bp.u[0] = *(const v4u*)(vp);
            bp.u[1] = *(const v4u*)(vp + 8);
            macc[j] = wmma_bf16(at, bp.v, macc[j]);
        }
    }

    // reduce tau mass across the 16 lanes of each half
#pragma unroll
    for (int r = 0; r < 8; ++r) {
        float v = tausum[r];
        v += __shfl_xor(v, 1, 32);
        v += __shfl_xor(v, 2, 32);
        v += __shfl_xor(v, 4, 32);
        v += __shfl_xor(v, 8, 32);
        tausum[r] = fmaxf(v, 1e-6f);
    }

#pragma unroll
    for (int j = 0; j < 4; ++j)
#pragma unroll
        for (int r = 0; r < 8; ++r) {
            int t = t0 + r + hi * 8;
            float mv = macc[j][r] / tausum[r];
            msg[(size_t)t * DD + h * HD + j * 16 + l15] = f2bf(mv);
        }
}

// ---------------- output GEMM + residual ----------------
// y = x + msg @ wo ; msg bf16 [T][D], woT bf16 [D][D], x/y f32 [T][D]
__global__ void gemm_out_kernel(const unsigned short* __restrict__ msg,
                                const unsigned short* __restrict__ woT,
                                const float* __restrict__ x,
                                float* __restrict__ y) {
    int lane = threadIdx.x & 31;
    int l15 = lane & 15, hi = lane >> 4;
    int gw = blockIdx.x * 8 + (threadIdx.x >> 5);   // 2048 waves
    int mt = gw >> 4, ng = gw & 15;
    int m0 = mt * 16, n0 = ng * 64;

    v8f acc[4];
#pragma unroll
    for (int j = 0; j < 4; ++j) acc[j] = (v8f){0,0,0,0,0,0,0,0};

    const unsigned short* abase = msg + (size_t)m0 * DD;
    for (int k0 = 0; k0 < DD; k0 += 32) {
        __builtin_prefetch(abase + (size_t)l15 * DD + k0 + 64, 0, 1);
        v16bf a = loadA16x32(abase + k0, DD, l15, hi);
#pragma unroll
        for (int j = 0; j < 4; ++j) {
            v16bf b = loadBT16(woT + (size_t)(n0 + j * 16) * DD + k0, DD, l15, hi);
            acc[j] = wmma_bf16(a, b, acc[j]);
        }
    }
#pragma unroll
    for (int j = 0; j < 4; ++j)
#pragma unroll
        for (int r = 0; r < 8; ++r) {
            int t = m0 + r + hi * 8;
            size_t idx = (size_t)t * DD + n0 + j * 16 + l15;
            y[idx] = x[idx] + acc[j][r];
        }
}

// ---------------- RMS norm ----------------
__global__ void rms_kernel(const float* __restrict__ y,
                           const float* __restrict__ scale,
                           float* __restrict__ out) {
    __shared__ float red[8];
    int t = blockIdx.x;
    const float* row = y + (size_t)t * DD;
    float s = 0.0f;
    for (int i = threadIdx.x; i < DD; i += 256) {
        float v = row[i];
        s += v * v;
    }
    s += __shfl_xor(s, 1, 32);
    s += __shfl_xor(s, 2, 32);
    s += __shfl_xor(s, 4, 32);
    s += __shfl_xor(s, 8, 32);
    s += __shfl_xor(s, 16, 32);
    if ((threadIdx.x & 31) == 0) red[threadIdx.x >> 5] = s;
    __syncthreads();
    float tot = red[0];
#pragma unroll
    for (int i = 1; i < 8; ++i) tot += red[i];
    float r = rsqrtf(tot * (1.0f / (float)DD) + 1e-6f);
    for (int i = threadIdx.x; i < DD; i += 256)
        out[(size_t)t * DD + i] = row[i] * r * scale[i];
}

// ---------------- launch ----------------
extern "C" void kernel_launch(void* const* d_in, const int* in_sizes, int n_in,
                              void* d_out, int out_size, void* d_ws, size_t ws_size,
                              hipStream_t stream) {
    (void)in_sizes; (void)n_in; (void)out_size; (void)ws_size;
    const float* x    = (const float*)d_in[0];
    const float* wq   = (const float*)d_in[1];
    const float* wk   = (const float*)d_in[2];
    const float* wv   = (const float*)d_in[3];
    const float* wo   = (const float*)d_in[4];
    const float* nsc  = (const float*)d_in[5];

    char* ws = (char*)d_ws;
    unsigned short* xbf   = (unsigned short*)(ws + (size_t)0);          // 4 MB
    unsigned short* wcatT = (unsigned short*)(ws + ((size_t)4  << 20)); // 6 MB
    unsigned short* woT   = (unsigned short*)(ws + ((size_t)10 << 20)); // 2 MB
    unsigned short* qb    = (unsigned short*)(ws + ((size_t)12 << 20)); // 4 MB
    unsigned short* kb    = (unsigned short*)(ws + ((size_t)16 << 20)); // 4 MB
    unsigned short* vT    = (unsigned short*)(ws + ((size_t)20 << 20)); // 4 MB
    unsigned short* msg   = (unsigned short*)(ws + ((size_t)24 << 20)); // 4 MB
    float*          ybuf  = (float*)         (ws + ((size_t)28 << 20)); // 8 MB

    cvt_bf16_kernel<<<(TT * DD) / 256, 256, 0, stream>>>(x, xbf, TT * DD);

    dim3 tb(32, 8), tg(32, 32);
    transposeW_kernel<<<tg, tb, 0, stream>>>(wq, wcatT);
    transposeW_kernel<<<tg, tb, 0, stream>>>(wk, wcatT + ((size_t)1 << 20));
    transposeW_kernel<<<tg, tb, 0, stream>>>(wv, wcatT + ((size_t)2 << 20));
    transposeW_kernel<<<tg, tb, 0, stream>>>(wo, woT);

    gemm_qkv_kernel<<<768, 256, 0, stream>>>(xbf, wcatT, qb, kb, vT);
    attn_kernel<<<256, 256, 0, stream>>>(qb, kb, vT, msg);
    gemm_out_kernel<<<256, 256, 0, stream>>>(msg, woT, x, ybuf);
    rms_kernel<<<TT, 256, 0, stream>>>(ybuf, nsc, (float*)d_out);
}
